// GAT_49246095016405
// MI455X (gfx1250) — matrix operand order repaired and checked by reading, not compiled, not run on hardware
//
#include <hip/hip_runtime.h>

typedef __attribute__((ext_vector_type(16))) _Float16 v16h;
typedef __attribute__((ext_vector_type(8)))  _Float16 v8h;
typedef __attribute__((ext_vector_type(8)))  float    v8f;

#define NEG_SLOPE 0.2f
#define EPSQ 1e-5f

__device__ __forceinline__ unsigned flipf(float f) {
  unsigned u = __float_as_uint(f);
  return (u & 0x80000000u) ? ~u : (u | 0x80000000u);
}
__device__ __forceinline__ float unflipf(unsigned u) {
  return (u & 0x80000000u) ? __uint_as_float(u ^ 0x80000000u)
                           : __uint_as_float(~u);
}

// ---------------- degree / self-loop edge attr ----------------
__global__ void k_deg(const int* __restrict__ dst, const float* __restrict__ ea,
                      float* __restrict__ cnt, float* __restrict__ asum, int E) {
  int e = blockIdx.x * blockDim.x + threadIdx.x;
  if (e >= E) return;
  int d = dst[e];
  atomicAdd(cnt + d, 1.0f);
  atomicAdd(asum + d, ea[e]);
}

__global__ void k_loopattr(const float* __restrict__ cnt, const float* __restrict__ asum,
                           float* __restrict__ la, int N) {
  int i = blockIdx.x * blockDim.x + threadIdx.x;
  if (i >= N) return;
  la[i] = asum[i] / fmaxf(cnt[i], 1.0f);
}

// ---------------- f32 -> f16 cast ----------------
__global__ void k_cast_f16(const float* __restrict__ x, _Float16* __restrict__ y, int n) {
  int i = blockIdx.x * blockDim.x + threadIdx.x;
  if (i < n) y[i] = (_Float16)x[i];
}

// ---------------- weight pack: W[Nout,K] f32 -> WMMA B-fragment order f16 ----
// Layout: Bp[(((kt*NT + nt)*32 + lane)*16 + e)]; NT = Nout/16.
// B element for lane: column n = nt*16 + (lane&15); K-row = kt*32 + e + (lane>=16 ? 16 : 0).
__global__ void k_pack_w(const float* __restrict__ W, _Float16* __restrict__ Bp,
                         int K, int Nout) {
  int id = blockIdx.x * blockDim.x + threadIdx.x;
  if (id >= K * Nout) return;
  int e    = id & 15;
  int lane = (id >> 4) & 31;
  int rest = id >> 9;
  int NT   = Nout >> 4;
  int nt   = rest % NT;
  int kt   = rest / NT;
  int n  = (nt << 4) + (lane & 15);
  int kk = (kt << 5) + e + ((lane & 16) ? 16 : 0);
  Bp[id] = (_Float16)W[(size_t)n * K + kk];
}

// ---------------- WMMA GEMM: Y[M,Nout] = A[M,K](f16) * Bp + bias --------------
// Block = 64 threads (2 waves); each wave computes a 16x64 strip.
// grid = (M/16, Nout/128). M % 16 == 0, Nout % 128 == 0.
__global__ void k_wmma_gemm(const _Float16* __restrict__ A,
                            const _Float16* __restrict__ Bp,
                            const float* __restrict__ bias,
                            float* __restrict__ Y,
                            int M, int K, int Nout) {
  const int lane   = threadIdx.x & 31;
  const int wave   = threadIdx.x >> 5;
  const int lane16 = lane & 15;
  const int hi8    = (lane & 16) ? 8 : 0;
  const int mBase  = blockIdx.x * 16;
  const int nBase  = (blockIdx.y * 2 + wave) * 64;
  const int NT     = Nout >> 4;

  const v8f vzero = {0.f, 0.f, 0.f, 0.f, 0.f, 0.f, 0.f, 0.f};
  v8f acc[4];
  acc[0] = vzero; acc[1] = vzero; acc[2] = vzero; acc[3] = vzero;

  const _Float16* arow = A + (size_t)(mBase + lane16) * K;

  for (int k = 0; k < K; k += 32) {
    // A 16x32 fragment in this code's k order (same as Bp, not the ISA table order): per lane one contiguous 32B run, k = 16*(lane/16) + e.
    v8h alo = *(const v8h*)(arow + k + 2 * hi8);
    v8h ahi = *(const v8h*)(arow + k + 2 * hi8 + 8);
    v16h a;
#pragma unroll
    for (int i = 0; i < 8; i++) { a[i] = alo[i]; a[8 + i] = ahi[i]; }

    const _Float16* bbase = Bp + ((size_t)(k >> 5) * NT) * 512 + (size_t)lane * 16;
#pragma unroll
    for (int t = 0; t < 4; t++) {
      v16h b = *(const v16h*)(bbase + (size_t)((nBase >> 4) + t) * 512);
      acc[t] = __builtin_amdgcn_wmma_f32_16x16x32_f16(
          false, a, false, b, (short)0, acc[t], false, false);
    }
  }

#pragma unroll
  for (int t = 0; t < 4; t++) {
    int nc = nBase + t * 16 + lane16;
    float bv = bias[nc];
#pragma unroll
    for (int r = 0; r < 8; r++) {
      int m = mBase + r + hi8;                 // C/D layout: lanes>=16 hold M=r+8
      Y[(size_t)m * Nout + nc] = acc[t][r] + bv;
    }
  }
}

// ---------------- edge attention logits + segment max (wave per edge) --------
__global__ void k_edge_logits(const float* __restrict__ xl, const float* __restrict__ xr,
                              const float* __restrict__ We, const float* __restrict__ att,
                              const int* __restrict__ src, const int* __restrict__ dst,
                              const float* __restrict__ eattr, const float* __restrict__ la,
                              float* __restrict__ logits, unsigned* __restrict__ mflip,
                              int E, int N, int D, int C) {
  int lane = threadIdx.x & 31;
  int e = blockIdx.x * (blockDim.x >> 5) + (threadIdx.x >> 5);
  int Etot = E + N;
  if (e >= Etot) return;
  int s, d; float ea;
  if (e < E) { s = src[e]; d = dst[e]; ea = eattr[e]; }
  else       { s = d = e - E; ea = la[s]; }
  const float* pl = xl + (size_t)s * D;
  const float* pr = xr + (size_t)d * D;
  float a0 = 0.f, a1 = 0.f;
  for (int c = lane; c < C; c += 32) {
    float v0 = pl[c] + pr[c] + ea * We[c];
    v0 = (v0 > 0.f) ? v0 : v0 * NEG_SLOPE;
    a0 += v0 * att[c];
    int c1 = c + C;
    float v1 = pl[c1] + pr[c1] + ea * We[c1];
    v1 = (v1 > 0.f) ? v1 : v1 * NEG_SLOPE;
    a1 += v1 * att[c1];
  }
#pragma unroll
  for (int off = 16; off > 0; off >>= 1) {
    a0 += __shfl_xor(a0, off, 32);
    a1 += __shfl_xor(a1, off, 32);
  }
  if (lane == 0) {
    logits[(size_t)e * 2 + 0] = a0;
    logits[(size_t)e * 2 + 1] = a1;
    atomicMax(mflip + (size_t)d * 2 + 0, flipf(a0));
    atomicMax(mflip + (size_t)d * 2 + 1, flipf(a1));
  }
}

// ---------------- softmax numerator p = exp(l - m[dst]); denom += p ----------
__global__ void k_softmax_p(const int* __restrict__ dst, float* __restrict__ logits,
                            const unsigned* __restrict__ mflip, float* __restrict__ denom,
                            int E, int N) {
  int id = blockIdx.x * blockDim.x + threadIdx.x;
  int tot = (E + N) * 2;
  if (id >= tot) return;
  int e = id >> 1, h = id & 1;
  int d = (e < E) ? dst[e] : (e - E);
  float m = unflipf(mflip[(size_t)d * 2 + h]);
  float p = __expf(logits[id] - m);
  logits[id] = p;
  atomicAdd(denom + (size_t)d * 2 + h, p);
}

__global__ void k_alpha(const int* __restrict__ dst, float* __restrict__ p,
                        const float* __restrict__ denom, int E, int N) {
  int id = blockIdx.x * blockDim.x + threadIdx.x;
  int tot = (E + N) * 2;
  if (id >= tot) return;
  int e = id >> 1, h = id & 1;
  int d = (e < E) ? dst[e] : (e - E);
  p[id] = p[id] / denom[(size_t)d * 2 + h];
}

// ---------------- weighted scatter-add of messages (block per edge) ----------
__global__ void k_scatter(const int* __restrict__ src, const int* __restrict__ dst,
                          const float* __restrict__ alpha, const float* __restrict__ xl,
                          float* __restrict__ out, int E, int N, int D, int C) {
  int e = blockIdx.x;
  int s, d;
  if (e < E) { s = src[e]; d = dst[e]; }
  else       { s = d = e - E; }
  float a0 = alpha[(size_t)e * 2 + 0];
  float a1 = alpha[(size_t)e * 2 + 1];
  const float* pl = xl + (size_t)s * D;
  float* po = out + (size_t)d * D;
  for (int c = threadIdx.x; c < D; c += blockDim.x) {
    float a = (c < C) ? a0 : a1;
    atomicAdd(po + c, a * pl[c]);
  }
}

// ---------------- bias + SiLU + RMSNorm -> f16 (D=512, wave per node) --------
__global__ void k_silu_rms_f16(const float* __restrict__ agg, const float* __restrict__ bias,
                               const float* __restrict__ w, _Float16* __restrict__ outh,
                               int N) {
  const int D = 512;
  int lane = threadIdx.x & 31;
  int n = blockIdx.x * (blockDim.x >> 5) + (threadIdx.x >> 5);
  if (n >= N) return;
  const float* pa = agg + (size_t)n * D;
  float vals[16];
  float ss = 0.f;
#pragma unroll
  for (int i = 0; i < 16; i++) {
    int c = lane + i * 32;
    float t = pa[c] + bias[c];
    float s = t / (1.f + __expf(-t));   // silu
    vals[i] = s;
    ss += s * s;
  }
#pragma unroll
  for (int off = 16; off > 0; off >>= 1) ss += __shfl_xor(ss, off, 32);
  float r = rsqrtf(ss * (1.0f / D) + EPSQ);
  _Float16* po = outh + (size_t)n * D;
#pragma unroll
  for (int i = 0; i < 16; i++) {
    int c = lane + i * 32;
    po[c] = (_Float16)(vals[i] * r * w[c]);
  }
}

// ---------------- bias + RMSNorm -> f32 (D=128, wave per node) ---------------
__global__ void k_rms_f32(const float* __restrict__ agg, const float* __restrict__ bias,
                          const float* __restrict__ w, float* __restrict__ out, int N) {
  const int D = 128;
  int lane = threadIdx.x & 31;
  int n = blockIdx.x * (blockDim.x >> 5) + (threadIdx.x >> 5);
  if (n >= N) return;
  const float* pa = agg + (size_t)n * D;
  float vals[4];
  float ss = 0.f;
#pragma unroll
  for (int i = 0; i < 4; i++) {
    int c = lane + i * 32;
    float t = pa[c] + bias[c];
    vals[i] = t;
    ss += t * t;
  }
#pragma unroll
  for (int off = 16; off > 0; off >>= 1) ss += __shfl_xor(ss, off, 32);
  float r = rsqrtf(ss * (1.0f / D) + EPSQ);
  float* po = out + (size_t)n * D;
#pragma unroll
  for (int i = 0; i < 4; i++) {
    int c = lane + i * 32;
    po[c] = vals[i] * r * w[c];
  }
}

// ---------------- final classifier: [N,128] x [20,128]^T --------------------
__global__ void k_out_linear(const float* __restrict__ h, const float* __restrict__ W,
                             float* __restrict__ out, int N, int K, int CL) {
  int id = blockIdx.x * blockDim.x + threadIdx.x;
  if (id >= N * CL) return;
  int n = id / CL, j = id % CL;
  const float* ph = h + (size_t)n * K;
  const float* pw = W + (size_t)j * K;
  float s = 0.f;
  for (int k = 0; k < K; k++) s += ph[k] * pw[k];
  out[id] = s;
}

// =============================================================================
extern "C" void kernel_launch(void* const* d_in, const int* in_sizes, int n_in,
                              void* d_out, int out_size, void* d_ws, size_t ws_size,
                              hipStream_t stream) {
  const float* x     = (const float*)d_in[0];
  const int*   eidx  = (const int*)d_in[1];
  const float* eattr = (const float*)d_in[2];
  const float* Wl1 = (const float*)d_in[3];  const float* bl1 = (const float*)d_in[4];
  const float* Wr1 = (const float*)d_in[5];  const float* br1 = (const float*)d_in[6];
  const float* We1 = (const float*)d_in[7];  const float* att1 = (const float*)d_in[8];
  const float* bias1 = (const float*)d_in[9];
  const float* Wl2 = (const float*)d_in[10]; const float* bl2 = (const float*)d_in[11];
  const float* Wr2 = (const float*)d_in[12]; const float* br2 = (const float*)d_in[13];
  const float* We2 = (const float*)d_in[14]; const float* att2 = (const float*)d_in[15];
  const float* bias2 = (const float*)d_in[16];
  const float* wln1 = (const float*)d_in[17];
  const float* wln3 = (const float*)d_in[18];
  const float* Wout = (const float*)d_in[19];

  const int DIM = 128, D1 = 512, D2 = 128, CL = 20;
  const int N = in_sizes[0] / DIM;
  const int E = in_sizes[1] / 2;
  const int Etot = E + N;
  const int* src = eidx;
  const int* dst = eidx + E;

  // ---- workspace layout (256B aligned chunks) ----
  char* wsb = (char*)d_ws;
  size_t off = 0;
  auto take = [&](size_t bytes) -> char* {
    char* p = wsb + off;
    off += (bytes + 255) & ~(size_t)255;
    return p;
  };
  float*    cnt   = (float*)take((size_t)2 * N * 4);           // cnt | asum
  float*    asum  = cnt + N;
  float*    la    = (float*)take((size_t)N * 4);               // loop_attr
  unsigned* mflip = (unsigned*)take((size_t)4 * N * 4);        // mflip(2N) | denom(2N)
  float*    denom = (float*)(mflip + (size_t)2 * N);
  _Float16* wpack = (_Float16*)take((size_t)512 * 512 * 2);    // packed weights (reused)
  float*    xl1   = (float*)take((size_t)N * D1 * 4);
  float*    xr1   = (float*)take((size_t)N * D1 * 4);          // reused as agg1
  float*    agg1  = xr1;
  float*    logits = (float*)take((size_t)Etot * 2 * 4);       // reused both layers
  _Float16* h1h   = (_Float16*)take((size_t)N * D1 * 2);       // also x-f16 staging
  _Float16* xh    = h1h;                                       // N*128 f16 fits inside
  // layer-2 tensors carved from dead xl1 region (4 x N*128 f32 = N*512 f32)
  float* xl2  = xl1;
  float* xr2  = xl1 + (size_t)N * 128;
  float* agg2 = xl1 + (size_t)N * 256;
  float* h2   = xl1 + (size_t)N * 384;

  auto cdiv = [](int a, int b) { return (a + b - 1) / b; };

  // ===== self-loop attrs =====
  hipMemsetAsync(cnt, 0, (size_t)2 * N * 4, stream);
  k_deg<<<cdiv(E, 256), 256, 0, stream>>>(dst, eattr, cnt, asum, E);
  k_loopattr<<<cdiv(N, 256), 256, 0, stream>>>(cnt, asum, la, N);

  // ===== layer 1 GEMMs (WMMA) =====
  k_cast_f16<<<cdiv(N * DIM, 256), 256, 0, stream>>>(x, xh, N * DIM);
  k_pack_w<<<cdiv(DIM * D1, 256), 256, 0, stream>>>(Wl1, wpack, DIM, D1);
  k_wmma_gemm<<<dim3(N / 16, D1 / 128), 64, 0, stream>>>(xh, wpack, bl1, xl1, N, DIM, D1);
  k_pack_w<<<cdiv(DIM * D1, 256), 256, 0, stream>>>(Wr1, wpack, DIM, D1);
  k_wmma_gemm<<<dim3(N / 16, D1 / 128), 64, 0, stream>>>(xh, wpack, br1, xr1, N, DIM, D1);

  // ===== layer 1 attention =====
  hipMemsetAsync(mflip, 0, (size_t)4 * N * 4, stream);
  k_edge_logits<<<cdiv(Etot, 8), 256, 0, stream>>>(xl1, xr1, We1, att1, src, dst, eattr,
                                                   la, logits, mflip, E, N, D1, 256);
  hipMemsetAsync(agg1, 0, (size_t)N * D1 * 4, stream);     // xr1 dead now
  k_softmax_p<<<cdiv(Etot * 2, 256), 256, 0, stream>>>(dst, logits, mflip, denom, E, N);
  k_alpha<<<cdiv(Etot * 2, 256), 256, 0, stream>>>(dst, logits, denom, E, N);
  k_scatter<<<Etot, 256, 0, stream>>>(src, dst, logits, xl1, agg1, E, N, D1, 256);

  // ===== bias + SiLU + RMSNorm -> h1 (f16) =====
  k_silu_rms_f16<<<cdiv(N, 8), 256, 0, stream>>>(agg1, bias1, wln1, h1h, N);

  // ===== layer 2 GEMMs (WMMA) =====
  k_pack_w<<<cdiv(D1 * D2, 256), 256, 0, stream>>>(Wl2, wpack, D1, D2);
  k_wmma_gemm<<<dim3(N / 16, D2 / 128), 64, 0, stream>>>(h1h, wpack, bl2, xl2, N, D1, D2);
  k_pack_w<<<cdiv(D1 * D2, 256), 256, 0, stream>>>(Wr2, wpack, D1, D2);
  k_wmma_gemm<<<dim3(N / 16, D2 / 128), 64, 0, stream>>>(h1h, wpack, br2, xr2, N, D1, D2);

  // ===== layer 2 attention =====
  hipMemsetAsync(mflip, 0, (size_t)4 * N * 4, stream);
  k_edge_logits<<<cdiv(Etot, 8), 256, 0, stream>>>(xl2, xr2, We2, att2, src, dst, eattr,
                                                   la, logits, mflip, E, N, D2, 64);
  hipMemsetAsync(agg2, 0, (size_t)N * D2 * 4, stream);
  k_softmax_p<<<cdiv(Etot * 2, 256), 256, 0, stream>>>(dst, logits, mflip, denom, E, N);
  k_alpha<<<cdiv(Etot * 2, 256), 256, 0, stream>>>(dst, logits, denom, E, N);
  k_scatter<<<Etot, 128, 0, stream>>>(src, dst, logits, xl2, agg2, E, N, D2, 64);

  // ===== bias + RMSNorm -> h2 (f32), classifier =====
  k_rms_f32<<<cdiv(N, 8), 256, 0, stream>>>(agg2, bias2, wln3, h2, N);
  k_out_linear<<<cdiv(N * CL, 256), 256, 0, stream>>>(h2, Wout, (float*)d_out, N, D2, CL);
}
